// GAEEncoder_3710851743894
// MI455X (gfx1250) — compile-verified, matrix operand order checked by reading, not run on hardware
//
#include <hip/hip_runtime.h>
#include <math.h>

// ---------------------------------------------------------------------------
// Types for CDNA5 WMMA (gfx1250, wave32)
// ---------------------------------------------------------------------------
typedef __attribute__((ext_vector_type(16))) __bf16 v16bf;
typedef __attribute__((ext_vector_type(8)))  float  v8f;

__device__ __forceinline__ unsigned short f2bf_rne(float x) {
    union { float f; unsigned u; } cv;
    cv.f = x;
    unsigned u = cv.u;
    u += 0x7FFFu + ((u >> 16) & 1u);   // round-to-nearest-even
    return (unsigned short)(u >> 16);
}

// ---------------------------------------------------------------------------
// Kernel: per-edge weighted degree sums + plain degree counts (atomics)
// ---------------------------------------------------------------------------
__global__ void edge_stats_kernel(const float* __restrict__ ew,
                                  const int* __restrict__ src,
                                  const int* __restrict__ dst,
                                  float* __restrict__ ows, float* __restrict__ iws,
                                  float* __restrict__ odeg, float* __restrict__ ideg,
                                  int E) {
    int e = blockIdx.x * blockDim.x + threadIdx.x;
    if (e >= E) return;
    float w = ew[e];
    int s = src[e], d = dst[e];
    atomicAdd(&ows[s], w);
    atomicAdd(&iws[d], w);
    atomicAdd(&odeg[s], 1.0f);
    atomicAdd(&ideg[d], 1.0f);
}

// ---------------------------------------------------------------------------
// Kernel: EdgeWeightNorm(norm='both'): nw = w * rsqrt(ows[s]) * rsqrt(iws[d])
// ---------------------------------------------------------------------------
__global__ void edge_norm_kernel(const float* __restrict__ ew,
                                 const int* __restrict__ src,
                                 const int* __restrict__ dst,
                                 const float* __restrict__ ows,
                                 const float* __restrict__ iws,
                                 float* __restrict__ nw, int E) {
    int e = blockIdx.x * blockDim.x + threadIdx.x;
    if (e >= E) return;
    nw[e] = ew[e] * rsqrtf(ows[src[e]]) * rsqrtf(iws[dst[e]]);
}

// ---------------------------------------------------------------------------
// Pack A (fp32 [M,K], optional per-row rsqrt(max(deg,1)) scale) into bf16
// fragments: layout [rowTile][kTile][lane][16 bf16]  (32B contiguous per lane).
// A-frag element map (CDNA5 ISA 7.12.2): lane r=lane&15 -> row, h=lane>>4;
// elems 0..7 -> K=kTile*32 + h*8 + e ; elems 8..15 -> K=kTile*32 +16+ h*8 + e.
// ---------------------------------------------------------------------------
__global__ void pack_a_bf16(const float* __restrict__ A,
                            const float* __restrict__ rowdeg,
                            unsigned short* __restrict__ Ap,
                            int M, int K) {
    const int nKT = K >> 5;
    int idx = blockIdx.x * blockDim.x + threadIdx.x;
    int total = (M >> 4) * nKT * 32;
    if (idx >= total) return;
    int lane = idx & 31;
    int frag = idx >> 5;
    int rowTile = frag / nKT;
    int kTile   = frag - rowTile * nKT;
    int r = lane & 15, h = lane >> 4;
    int row = rowTile * 16 + r;
    float sc = rowdeg ? rsqrtf(fmaxf(rowdeg[row], 1.0f)) : 1.0f;
    const float* arow = A + (size_t)row * K + kTile * 32;
    float4 a0 = *(const float4*)(arow + h * 8);
    float4 a1 = *(const float4*)(arow + h * 8 + 4);
    float4 a2 = *(const float4*)(arow + 16 + h * 8);
    float4 a3 = *(const float4*)(arow + 16 + h * 8 + 4);
    union { unsigned short s[16]; uint4 q[2]; } u;
    u.s[0]  = f2bf_rne(a0.x * sc); u.s[1]  = f2bf_rne(a0.y * sc);
    u.s[2]  = f2bf_rne(a0.z * sc); u.s[3]  = f2bf_rne(a0.w * sc);
    u.s[4]  = f2bf_rne(a1.x * sc); u.s[5]  = f2bf_rne(a1.y * sc);
    u.s[6]  = f2bf_rne(a1.z * sc); u.s[7]  = f2bf_rne(a1.w * sc);
    u.s[8]  = f2bf_rne(a2.x * sc); u.s[9]  = f2bf_rne(a2.y * sc);
    u.s[10] = f2bf_rne(a2.z * sc); u.s[11] = f2bf_rne(a2.w * sc);
    u.s[12] = f2bf_rne(a3.x * sc); u.s[13] = f2bf_rne(a3.y * sc);
    u.s[14] = f2bf_rne(a3.z * sc); u.s[15] = f2bf_rne(a3.w * sc);
    uint4* o = (uint4*)(Ap + (size_t)idx * 16);
    o[0] = u.q[0];
    o[1] = u.q[1];
}

// ---------------------------------------------------------------------------
// Pack B (fp32 [K,Ncol], row-major) into bf16 fragments:
// layout [kTile][colTile][lane][16 bf16]; lane = output column colTile*16 + r,
// same K packing as A (symmetric B-fragment convention).
// ---------------------------------------------------------------------------
__global__ void pack_b_bf16(const float* __restrict__ B,
                            unsigned short* __restrict__ Bp,
                            int K, int Ncol) {
    const int nCT = Ncol >> 4;
    int idx = blockIdx.x * blockDim.x + threadIdx.x;
    int total = (K >> 5) * nCT * 32;
    if (idx >= total) return;
    int lane = idx & 31;
    int frag = idx >> 5;
    int kTile   = frag / nCT;
    int colTile = frag - kTile * nCT;
    int r = lane & 15, h = lane >> 4;
    const float* bcol = B + (size_t)(kTile * 32) * Ncol + colTile * 16 + r;
    union { unsigned short s[16]; uint4 q[2]; } u;
#pragma unroll
    for (int e = 0; e < 8; ++e)
        u.s[e] = f2bf_rne(bcol[(size_t)(h * 8 + e) * Ncol]);
#pragma unroll
    for (int e = 0; e < 8; ++e)
        u.s[8 + e] = f2bf_rne(bcol[(size_t)(16 + h * 8 + e) * Ncol]);
    uint4* o = (uint4*)(Bp + (size_t)idx * 16);
    o[0] = u.q[0];
    o[1] = u.q[1];
}

// ---------------------------------------------------------------------------
// WMMA GEMM on pre-packed bf16 fragments: C[M,Ncol] = A @ B (f32 accumulate).
// One wave computes a 16(M) x 64(N) strip: 4 tiles, K step 32.
// Inner loop: 2x b128 loads (A frag) + 8x b128 loads (4 B frags) + 4 WMMAs.
// ---------------------------------------------------------------------------
__global__ void wmma_gemm_packed(const v16bf* __restrict__ Ap,
                                 const v16bf* __restrict__ Bp,
                                 float* __restrict__ C,
                                 int M, int nKT, int Ncol) {
    const int lane   = threadIdx.x & 31;
    const int wave   = blockIdx.x * (blockDim.x >> 5) + (threadIdx.x >> 5);
    const int nChunk = Ncol >> 6;                 // 64-wide column strips
    const int rowBlk = wave / nChunk;
    if (rowBlk * 16 >= M) return;
    const int colChunk = wave - rowBlk * nChunk;
    const int nCT = Ncol >> 4;                    // 16-wide column tiles

    const v16bf* ap = Ap + ((size_t)rowBlk * nKT) * 32 + lane;
    const v16bf* bp = Bp + ((size_t)colChunk * 4) * 32 + lane;

    v8f acc0 = {}, acc1 = {}, acc2 = {}, acc3 = {};

#pragma unroll 2
    for (int kt = 0; kt < nKT; ++kt) {
        v16bf af = ap[(size_t)kt * 32];
        const v16bf* b = bp + (size_t)kt * nCT * 32;
        v16bf b0 = b[0];
        v16bf b1 = b[32];
        v16bf b2 = b[64];
        v16bf b3 = b[96];
        acc0 = __builtin_amdgcn_wmma_f32_16x16x32_bf16(false, af, false, b0, (short)0, acc0, false, false);
        acc1 = __builtin_amdgcn_wmma_f32_16x16x32_bf16(false, af, false, b1, (short)0, acc1, false, false);
        acc2 = __builtin_amdgcn_wmma_f32_16x16x32_bf16(false, af, false, b2, (short)0, acc2, false, false);
        acc3 = __builtin_amdgcn_wmma_f32_16x16x32_bf16(false, af, false, b3, (short)0, acc3, false, false);
    }

    // C/D layout: v8f elem i -> row = 8*(lane>>4) + i, col = lane&15
    const int r = lane & 15, h = lane >> 4;
    const int rowBase = rowBlk * 16, colBase = colChunk * 64;
#pragma unroll
    for (int i = 0; i < 8; ++i) {
        float* crow = C + (size_t)(rowBase + h * 8 + i) * Ncol + colBase;
        crow[r]      = acc0[i];
        crow[16 + r] = acc1[i];
        crow[32 + r] = acc2[i];
        crow[48 + r] = acc3[i];
    }
}

// ---------------------------------------------------------------------------
// SpMM scatter-add: agg[dst[e]] += w_e * h[src[e]]   (w_e = nw[e] or 1.0)
// ---------------------------------------------------------------------------
__global__ void spmm_scatter_kernel(const float* __restrict__ h,
                                    const float* __restrict__ nw,   // nullptr => 1.0
                                    const int* __restrict__ src,
                                    const int* __restrict__ dst,
                                    float* __restrict__ agg,
                                    int E, int F) {
    const int chunks = F >> 2;
    long long idx = (long long)blockIdx.x * blockDim.x + threadIdx.x;
    long long total = (long long)E * chunks;
    if (idx >= total) return;
    int e = (int)(idx / chunks);
    int c = (int)(idx % chunks);
    float w = nw ? nw[e] : 1.0f;
    int s = src[e], d = dst[e];
    const float4 v = ((const float4*)(h + (long long)s * F))[c];
    float* o = agg + (long long)d * F + (long long)c * 4;
    atomicAdd(o + 0, w * v.x);
    atomicAdd(o + 1, w * v.y);
    atomicAdd(o + 2, w * v.z);
    atomicAdd(o + 3, w * v.w);
}

// ---------------------------------------------------------------------------
// Bias + ReLU
// ---------------------------------------------------------------------------
__global__ void bias_relu_kernel(float* __restrict__ x, const float* __restrict__ b,
                                 long long total, int F) {
    long long idx = (long long)blockIdx.x * blockDim.x + threadIdx.x;
    if (idx >= total) return;
    x[idx] = fmaxf(x[idx] + b[(int)(idx % F)], 0.0f);
}

// ---------------------------------------------------------------------------
// Final dst-side norm + bias: out = out * rsqrt(max(in_deg,1)) + b2
// ---------------------------------------------------------------------------
__global__ void final_scale_kernel(float* __restrict__ out,
                                   const float* __restrict__ ideg,
                                   const float* __restrict__ b2,
                                   long long total, int F) {
    long long idx = (long long)blockIdx.x * blockDim.x + threadIdx.x;
    if (idx >= total) return;
    int row = (int)(idx / F);
    int col = (int)(idx % F);
    out[idx] = out[idx] * rsqrtf(fmaxf(ideg[row], 1.0f)) + b2[col];
}

// ---------------------------------------------------------------------------
// Host-side orchestration
// ---------------------------------------------------------------------------
extern "C" void kernel_launch(void* const* d_in, const int* in_sizes, int n_in,
                              void* d_out, int out_size, void* d_ws, size_t ws_size,
                              hipStream_t stream) {
    const float* features = (const float*)d_in[0];
    const float* ew       = (const float*)d_in[1];
    const int*   src      = (const int*)  d_in[2];
    const int*   dst      = (const int*)  d_in[3];
    const float* W1       = (const float*)d_in[4];
    const float* b1       = (const float*)d_in[5];
    const float* W2       = (const float*)d_in[6];
    const float* b2       = (const float*)d_in[7];

    const int E  = in_sizes[1];
    const int H1 = in_sizes[5];
    const int H2 = in_sizes[7];
    const int IN = in_sizes[4] / H1;
    const int Nn = in_sizes[0] / IN;          // nodes

    // ---- workspace layout (aliased lifetimes) ----
    char* p = (char*)d_ws;
    auto alloc = [&](size_t bytes) {
        char* r = p;
        p += (bytes + 255) & ~(size_t)255;
        return r;
    };
    float* out_wsum = (float*)alloc((size_t)Nn * 4);
    float* in_wsum  = (float*)alloc((size_t)Nn * 4);
    float* out_deg  = (float*)alloc((size_t)Nn * 4);
    float* in_deg   = (float*)alloc((size_t)Nn * 4);
    float* nw       = (float*)alloc((size_t)E * 4);
    size_t szA = (size_t)Nn * IN * 2;            // Ap1 (bf16)
    size_t szH2 = (size_t)Nn * H2 * 4;           // h2 (f32)
    char* regionA = alloc(szA > szH2 ? szA : szH2);      // Ap1, later h2
    char* regionB = alloc((size_t)Nn * H1 * 4);          // h1, later Ap2
    float* agg1   = (float*)alloc((size_t)Nn * H1 * 4);
    unsigned short* Bp1 = (unsigned short*)alloc((size_t)IN * H1 * 2);
    unsigned short* Bp2 = (unsigned short*)alloc((size_t)H1 * H2 * 2);

    unsigned short* Ap1 = (unsigned short*)regionA;   // live: pack_a1 -> gemm1
    float*          h2  = (float*)regionA;            // live: gemm2 -> spmm2
    float*          h1  = (float*)regionB;            // live: gemm1 -> spmm1
    unsigned short* Ap2 = (unsigned short*)regionB;   // live: pack_a2 -> gemm2
    float*          out = (float*)d_out;

    // zero-init accumulators (fresh every call; graph-capture safe)
    hipMemsetAsync(out_wsum, 0, sizeof(float) * (size_t)4 * Nn, stream);
    hipMemsetAsync(agg1, 0, sizeof(float) * (size_t)Nn * H1, stream);
    hipMemsetAsync(out,  0, sizeof(float) * (size_t)Nn * H2, stream);

    const int T = 256;

    // 1) degree / weighted-degree sums
    edge_stats_kernel<<<(E + T - 1) / T, T, 0, stream>>>(
        ew, src, dst, out_wsum, in_wsum, out_deg, in_deg, E);

    // 2) edge weight normalization
    edge_norm_kernel<<<(E + T - 1) / T, T, 0, stream>>>(
        ew, src, dst, out_wsum, in_wsum, nw, E);

    // 3) pack operands for GEMM1
    {
        int totalA = (Nn / 16) * (IN / 32) * 32;
        pack_a_bf16<<<(totalA + T - 1) / T, T, 0, stream>>>(features, nullptr, Ap1, Nn, IN);
        int totalB = (IN / 32) * (H1 / 16) * 32;
        pack_b_bf16<<<(totalB + T - 1) / T, T, 0, stream>>>(W1, Bp1, IN, H1);
    }

    // 4) GEMM1: h1 = features @ W1
    {
        int waves = (Nn / 16) * (H1 / 64);
        int blocks = (waves * 32 + T - 1) / T;
        wmma_gemm_packed<<<blocks, T, 0, stream>>>(
            (const v16bf*)Ap1, (const v16bf*)Bp1, h1, Nn, IN / 32, H1);
    }

    // 5) SpMM1: agg1[dst] += nw * h1[src]
    {
        long long total = (long long)E * (H1 >> 2);
        int blocks = (int)((total + T - 1) / T);
        spmm_scatter_kernel<<<blocks, T, 0, stream>>>(h1, nw, src, dst, agg1, E, H1);
    }

    // 6) bias + relu
    {
        long long total = (long long)Nn * H1;
        int blocks = (int)((total + T - 1) / T);
        bias_relu_kernel<<<blocks, T, 0, stream>>>(agg1, b1, total, H1);
    }

    // 7) pack operands for GEMM2 (row scale rsqrt(max(out_deg,1)) fused into A pack)
    {
        int totalA = (Nn / 16) * (H1 / 32) * 32;
        pack_a_bf16<<<(totalA + T - 1) / T, T, 0, stream>>>(agg1, out_deg, Ap2, Nn, H1);
        int totalB = (H1 / 32) * (H2 / 16) * 32;
        pack_b_bf16<<<(totalB + T - 1) / T, T, 0, stream>>>(W2, Bp2, H1, H2);
    }

    // 8) GEMM2: h2 = (scaled agg1) @ W2
    {
        int waves = (Nn / 16) * (H2 / 64);
        int blocks = (waves * 32 + T - 1) / T;
        wmma_gemm_packed<<<blocks, T, 0, stream>>>(
            (const v16bf*)Ap2, (const v16bf*)Bp2, h2, Nn, H1 / 32, H2);
    }

    // 9) SpMM2: out[dst] += h2[src]  (unweighted)
    {
        long long total = (long long)E * (H2 >> 2);
        int blocks = (int)((total + T - 1) / T);
        spmm_scatter_kernel<<<blocks, T, 0, stream>>>(h2, nullptr, src, dst, out, E, H2);
    }

    // 10) out = out * rsqrt(max(in_deg,1)) + b2
    {
        long long total = (long long)Nn * H2;
        int blocks = (int)((total + T - 1) / T);
        final_scale_kernel<<<blocks, T, 0, stream>>>(out, in_deg, b2, total, H2);
    }
}
